// LogSparseAttention_72473278152868
// MI455X (gfx1250) — matrix operand order repaired and checked by reading, not compiled, not run on hardware
//
#include <hip/hip_runtime.h>
#include <hip/hip_bf16.h>

// ---------------------------------------------------------------------------
// LogSparse attention, MI455X (gfx1250, wave32).
// Pipeline:
//   1) cvt_f32_bf16 : hidden + Wq/Wk/Wv  f32 -> bf16 (RNE)      (~40 MB, BW-bound)
//   2) qkv_gemm     : fused Q/K/V projection via v_wmma_f32_16x16x32_bf16,
//                     f32 accumulate, bias epilogue, writes [3,B,NH,S,D] f32.
//                     Each wave owns a 32(M)x64(N) strip: 2 A-frags x 4 B-frags
//                     -> 8 WMMAs per K-step, B fragments reused across 2 WMMAs
//                     (1.5 b128 loads per WMMA).
//   3) logsparse_attn: one wave per (b,h,query); <=12 keys; shfl-reduced dot
//                     products; stable softmax; weighted V sum.
// ---------------------------------------------------------------------------

typedef __attribute__((ext_vector_type(16))) __bf16 v16bf;
typedef __attribute__((ext_vector_type(8)))  float  v8f;

#define B_  4
#define S_  2048
#define H_  1024
#define NH_ 16
#define D_  64
#define M_  (B_ * S_)          // 8192 rows in the fused GEMM

// ---------------------------------------------------------------------------
// f32 -> bf16 round-to-nearest-even
// ---------------------------------------------------------------------------
__global__ void cvt_f32_bf16(const float* __restrict__ src,
                             unsigned short* __restrict__ dst, int n) {
    int i = blockIdx.x * blockDim.x + threadIdx.x;
    int stride = gridDim.x * blockDim.x;
    for (; i < n; i += stride) {
        unsigned u = __float_as_uint(src[i]);
        unsigned r = u + 0x7fffu + ((u >> 16) & 1u);   // RNE to bf16
        dst[i] = (unsigned short)(r >> 16);
    }
}

// Load 16 bf16 (two 16B chunks) straight into a WMMA fragment register set.
__device__ __forceinline__ v16bf load_frag2(const unsigned short* p0,
                                            const unsigned short* p1) {
    union { uint4 q[2]; v16bf v; } f;
    f.q[0] = *reinterpret_cast<const uint4*>(p0);   // global_load_b128
    f.q[1] = *reinterpret_cast<const uint4*>(p1);   // global_load_b128
    return f.v;
}

// ---------------------------------------------------------------------------
// Fused QKV projection: out[z, m, n] = sum_k A[m,k] * W_z[n,k] + bias_z[n]
//   grid = (M/256, H/64, 3); block = 256 (8 waves).
//   wave -> 32(M) x 64(N) strip; output stored as [z][b][h][s][d] f32.
// ---------------------------------------------------------------------------
__global__ __launch_bounds__(256)
void qkv_gemm(const unsigned short* __restrict__ Abf,   // [M_, H_] bf16
              const unsigned short* __restrict__ Wbf,   // [3, H_, H_] bf16
              const float* __restrict__ bq,
              const float* __restrict__ bk,
              const float* __restrict__ bv,
              float* __restrict__ qkv)                  // [3, B_, NH_, S_, D_]
{
    const int lane = threadIdx.x & 31;
    const int wave = threadIdx.x >> 5;
    const int half = lane >> 4;      // lane group 0..15 / 16..31
    const int lrow = lane & 15;
    const int z    = blockIdx.z;
    const int m0   = (blockIdx.x * 8 + wave) * 32;
    const int n0   = blockIdx.y * 64;

    const unsigned short* W    = Wbf + (size_t)z * H_ * H_;
    const float*          bias = (z == 0) ? bq : ((z == 1) ? bk : bv);
    float*                obuf = qkv + (size_t)z * M_ * H_;

    // A 16x32 bf16 layout: lanes 0-15 hold K 0..15 of row
    // m+lrow; lanes 16-31 hold K 16..31.
    const unsigned short* ar0 = Abf + (size_t)(m0 +  0 + lrow) * H_ + half * 16;
    const unsigned short* ar1 = Abf + (size_t)(m0 + 16 + lrow) * H_ + half * 16;
    // B 32x16 bf16: lane = column n, lanes 0-15 hold K 0..15, lanes 16-31 hold
    // K 16..31 -> 16 contiguous bf16 from weight row n (reference uses W.T).
    const unsigned short* w0 = W + (size_t)(n0 +  0 + lrow) * H_ + half * 16;
    const unsigned short* w1 = W + (size_t)(n0 + 16 + lrow) * H_ + half * 16;
    const unsigned short* w2 = W + (size_t)(n0 + 32 + lrow) * H_ + half * 16;
    const unsigned short* w3 = W + (size_t)(n0 + 48 + lrow) * H_ + half * 16;

    v8f acc[8] = {};   // acc[t*2 + mi]: n-subtile t (0..3), m-subtile mi (0..1)

#pragma unroll 2
    for (int k0 = 0; k0 < H_; k0 += 32) {
        v16bf a0 = load_frag2(ar0 + k0, ar0 + k0 + 8);
        v16bf a1 = load_frag2(ar1 + k0, ar1 + k0 + 8);
        v16bf b0 = load_frag2(w0 + k0, w0 + k0 + 8);
        v16bf b1 = load_frag2(w1 + k0, w1 + k0 + 8);
        v16bf b2 = load_frag2(w2 + k0, w2 + k0 + 8);
        v16bf b3 = load_frag2(w3 + k0, w3 + k0 + 8);
        acc[0] = __builtin_amdgcn_wmma_f32_16x16x32_bf16(false, a0, false, b0,
                                                         (short)0, acc[0], false, false);
        acc[1] = __builtin_amdgcn_wmma_f32_16x16x32_bf16(false, a1, false, b0,
                                                         (short)0, acc[1], false, false);
        acc[2] = __builtin_amdgcn_wmma_f32_16x16x32_bf16(false, a0, false, b1,
                                                         (short)0, acc[2], false, false);
        acc[3] = __builtin_amdgcn_wmma_f32_16x16x32_bf16(false, a1, false, b1,
                                                         (short)0, acc[3], false, false);
        acc[4] = __builtin_amdgcn_wmma_f32_16x16x32_bf16(false, a0, false, b2,
                                                         (short)0, acc[4], false, false);
        acc[5] = __builtin_amdgcn_wmma_f32_16x16x32_bf16(false, a1, false, b2,
                                                         (short)0, acc[5], false, false);
        acc[6] = __builtin_amdgcn_wmma_f32_16x16x32_bf16(false, a0, false, b3,
                                                         (short)0, acc[6], false, false);
        acc[7] = __builtin_amdgcn_wmma_f32_16x16x32_bf16(false, a1, false, b3,
                                                         (short)0, acc[7], false, false);
    }

    // Epilogue. C/D layout: VGPR r -> M = mbase + half*8 + r, N = n_sub + lrow.
#pragma unroll
    for (int t = 0; t < 4; ++t) {
        const int n  = n0 + t * 16 + lrow;
        const float bn = bias[n];
        const int h  = n >> 6;        // head
        const int d  = n & (D_ - 1);  // dim within head
#pragma unroll
        for (int mi = 0; mi < 2; ++mi) {
#pragma unroll
            for (int r = 0; r < 8; ++r) {
                const int m  = m0 + mi * 16 + half * 8 + r;
                const int bb = m >> 11;          // m / S_
                const int ss = m & (S_ - 1);
                obuf[(((size_t)bb * NH_ + h) * S_ + ss) * D_ + d] =
                    acc[t * 2 + mi][r] + bn;
            }
        }
    }
}

// ---------------------------------------------------------------------------
// Log-sparse attention. One wave per (b, h, i). Keys: j = i and j = i - 2^k,
// k = 0..10 (diff <= 1024 for S=2048) -> <= 12 keys. Each lane owns 2 dims of
// the 64-dim head; dot products reduced with 5x shfl_xor (wave32).
// ---------------------------------------------------------------------------
__global__ __launch_bounds__(256)
void logsparse_attn(const float* __restrict__ qkv,    // [3, B_, NH_, S_, D_]
                    const float* __restrict__ mask,   // [B_, 1, 1, S_]
                    float* __restrict__ out)          // [B_, S_, H_]
{
    const int lane = threadIdx.x & 31;
    const int w = blockIdx.x * (blockDim.x >> 5) + (threadIdx.x >> 5);
    const int i = w & (S_ - 1);
    const int h = (w >> 11) & (NH_ - 1);
    const int b = w >> 15;                 // / (S_*NH_)

    const size_t bh = ((size_t)b * NH_ + h) * S_;
    const float* Qb = qkv + (bh + i) * D_;
    const float* Kb = qkv + (size_t)1 * M_ * H_ + bh * D_;
    const float* Vb = qkv + (size_t)2 * M_ * H_ + bh * D_;

    const int d0 = lane * 2;
    const float2 q2 = *reinterpret_cast<const float2*>(Qb + d0);
    const float scale = 0.125f;            // 1/sqrt(64)

    float sc[12];
    float smax = -3.0e38f;
#pragma unroll
    for (int t = 0; t < 12; ++t) {
        const int diff = (t == 0) ? 0 : (1 << (t - 1));
        const bool valid = diff <= i;      // uniform across the wave
        float s = -3.0e38f;
        if (valid) {
            const int j = i - diff;
            const float2 k2 = *reinterpret_cast<const float2*>(Kb + (size_t)j * D_ + d0);
            float p = q2.x * k2.x + q2.y * k2.y;
            p += __shfl_xor(p, 1);
            p += __shfl_xor(p, 2);
            p += __shfl_xor(p, 4);
            p += __shfl_xor(p, 8);
            p += __shfl_xor(p, 16);
            s = p * scale + mask[(size_t)b * S_ + j];
        }
        sc[t] = s;
        smax = fmaxf(smax, s);
    }

    float denom = 0.f;
#pragma unroll
    for (int t = 0; t < 12; ++t) {
        float e = __expf(sc[t] - smax);
        sc[t] = e;
        denom += e;
    }
    const float inv = 1.0f / denom;

    float2 acc = {0.f, 0.f};
#pragma unroll
    for (int t = 0; t < 12; ++t) {
        const int diff = (t == 0) ? 0 : (1 << (t - 1));
        if (diff <= i) {
            const int j = i - diff;
            const float2 v2 = *reinterpret_cast<const float2*>(Vb + (size_t)j * D_ + d0);
            acc.x += sc[t] * v2.x;
            acc.y += sc[t] * v2.y;
        }
    }
    acc.x *= inv;
    acc.y *= inv;

    // out layout [B, S, H]: heads re-interleaved
    *reinterpret_cast<float2*>(out + ((size_t)b * S_ + i) * H_ + h * D_ + d0) = acc;
}

// ---------------------------------------------------------------------------
extern "C" void kernel_launch(void* const* d_in, const int* in_sizes, int n_in,
                              void* d_out, int out_size, void* d_ws, size_t ws_size,
                              hipStream_t stream) {
    const float* hs   = (const float*)d_in[0];
    const float* mask = (const float*)d_in[1];
    const float* Wq   = (const float*)d_in[2];
    const float* bq   = (const float*)d_in[3];
    const float* Wk   = (const float*)d_in[4];
    const float* bk   = (const float*)d_in[5];
    const float* Wv   = (const float*)d_in[6];
    const float* bv   = (const float*)d_in[7];
    float* out = (float*)d_out;

    // Workspace layout (≈118 MB):
    //   [0)                bf16 hidden   M_*H_*2           = 16 MiB
    //   [16 MiB)           bf16 W q/k/v  3*H_*H_*2         =  6 MiB
    //   [22 MiB, aligned)  f32 Q/K/V     3*M_*H_*4         = 96 MiB
    char* ws = (char*)d_ws;
    unsigned short* Abf = (unsigned short*)ws;
    unsigned short* Wbf = (unsigned short*)(ws + (size_t)M_ * H_ * 2);
    float* qkv = (float*)(ws + (size_t)M_ * H_ * 2 + (size_t)3 * H_ * H_ * 2);

    cvt_f32_bf16<<<2048, 256, 0, stream>>>(hs, Abf, M_ * H_);
    cvt_f32_bf16<<<512, 256, 0, stream>>>(Wq, Wbf + (size_t)0 * H_ * H_, H_ * H_);
    cvt_f32_bf16<<<512, 256, 0, stream>>>(Wk, Wbf + (size_t)1 * H_ * H_, H_ * H_);
    cvt_f32_bf16<<<512, 256, 0, stream>>>(Wv, Wbf + (size_t)2 * H_ * H_, H_ * H_);

    dim3 g(M_ / 256, H_ / 64, 3);   // (32, 16, 3), 256 threads = 8 waves/block
    qkv_gemm<<<g, 256, 0, stream>>>(Abf, Wbf, bq, bk, bv, qkv);

    logsparse_attn<<<(B_ * NH_ * S_) / 8, 256, 0, stream>>>(qkv, mask, out);
}
